// SSAModule_86457691668722
// MI455X (gfx1250) — matrix operand order repaired and checked by reading, not compiled, not run on hardware
//
#include <hip/hip_runtime.h>
#include <hip/hip_bf16.h>

// ---------------------------------------------------------------------------
// Problem constants
// ---------------------------------------------------------------------------
#define BB 8
#define NN 256
#define HH 768
#define PP 128
#define NHEADS 12
#define DD 64
#define MROWS (BB * NN)   // 2048
#define EPS 1e-5f

typedef __attribute__((ext_vector_type(16))) __bf16 v16bf;
typedef __attribute__((ext_vector_type(8)))  float  v8f;

// ---------------------------------------------------------------------------
// WMMA helpers (CDNA5 wave32, 16x16x32 bf16 -> f32)
// ---------------------------------------------------------------------------
__device__ __forceinline__ v8f wmma_bf16(v16bf a, v16bf b, v8f c) {
  return __builtin_amdgcn_wmma_f32_16x16x32_bf16(
      /*neg_a=*/false, a, /*neg_b=*/false, b,
      /*c_mod=*/(short)0, c, /*reuse_a=*/false, /*reuse_b=*/false);
}

// A-fragment (16x32, MxK) from bf16 memory, row-major with leading dim ld.
// ISA layout: lanes 0-15 -> M=lane, K blocks {0..7, 16..23}; lanes 16-31 ->
// same M, K blocks {8..15, 24..31}; 2 K-values packed per VGPR.
__device__ __forceinline__ v16bf load_a_bf16(const __bf16* base, int ld, int lane) {
  int m  = lane & 15;
  int kg = lane >> 4;
  const __bf16* r = base + (size_t)m * ld;
  v16bf a;
#pragma unroll
  for (int j = 0; j < 8; ++j) {
    int k0 = (kg ? 8 : 0) + ((j & 4) ? 16 : 0) + 2 * (j & 3);
    a[2 * j]     = r[k0];
    a[2 * j + 1] = r[k0 + 1];
  }
  return a;
}

// Same but converting from f32 memory on the fly.
__device__ __forceinline__ v16bf load_a_f32(const float* base, int ld, int lane) {
  int m  = lane & 15;
  int kg = lane >> 4;
  const float* r = base + (size_t)m * ld;
  v16bf a;
#pragma unroll
  for (int j = 0; j < 8; ++j) {
    int k0 = (kg ? 8 : 0) + ((j & 4) ? 16 : 0) + 2 * (j & 3);
    a[2 * j]     = (__bf16)r[k0];
    a[2 * j + 1] = (__bf16)r[k0 + 1];
  }
  return a;
}

// B-fragment (32x16, KxN) where memory holds B TRANSPOSED: BT[n][k],
// row-major with leading dim ld. ISA layout: lane -> N = lane&15,
// lanes 0-15 K blocks {0..7, 16..23}, lanes 16-31 K blocks {8..15, 24..31}.
__device__ __forceinline__ v16bf load_bt_bf16(const __bf16* bt, int ld, int lane) {
  int n  = lane & 15;
  int kg = lane >> 4;
  const __bf16* r = bt + (size_t)n * ld;
  v16bf b;
#pragma unroll
  for (int j = 0; j < 8; ++j) {
    int k0 = (kg ? 8 : 0) + ((j & 4) ? 16 : 0) + 2 * (j & 3);
    b[2 * j]     = r[k0];
    b[2 * j + 1] = r[k0 + 1];
  }
  return b;
}

// A-fragment for the stacked-heads Q matrix (12 real rows, padded to 16).
__device__ __forceinline__ v16bf load_a_qheads(const __bf16* qh, int kk, int lane) {
  int m  = lane & 15;
  int kg = lane >> 4;
  v16bf a;
  if (m < NHEADS) {
    const __bf16* r = qh + m * DD + kk;
#pragma unroll
    for (int j = 0; j < 8; ++j) {
      int k0 = (kg ? 8 : 0) + ((j & 4) ? 16 : 0) + 2 * (j & 3);
      a[2 * j]     = r[k0];
      a[2 * j + 1] = r[k0 + 1];
    }
  } else {
#pragma unroll
    for (int i = 0; i < 16; ++i) a[i] = (__bf16)0.0f;
  }
  return a;
}

// ---------------------------------------------------------------------------
// K0: transpose f32 weight [K x N] -> bf16 WT [N x K]
// ---------------------------------------------------------------------------
__global__ void k_transpose_bf16(const float* __restrict__ W, __bf16* __restrict__ WT,
                                 int K, int N) {
  int idx = blockIdx.x * blockDim.x + threadIdx.x;
  if (idx < K * N) {
    int n = idx / K;
    int k = idx - n * K;
    WT[idx] = (__bf16)W[(size_t)k * N + n];
  }
}

// ---------------------------------------------------------------------------
// K1: X(2048x768) @ W(768x768) + b -> bf16 (q scaled; v stored transposed)
// grid (64, 12), block 256 (8 waves); each wave one 16x16 tile of a 32x64 block
// ---------------------------------------------------------------------------
__global__ __launch_bounds__(256) void k_gemm_qkv(
    const float* __restrict__ X, const __bf16* __restrict__ WT,
    const float* __restrict__ bvec, __bf16* __restrict__ dst,
    float scale, int vmode) {
  int lane = threadIdx.x & 31, w = threadIdx.x >> 5;
  int row0 = blockIdx.x * 32 + (w & 1) * 16;
  int col0 = blockIdx.y * 64 + (w >> 1) * 16;
  v8f acc = {};
#pragma unroll 4
  for (int kk = 0; kk < HH; kk += 32) {
    v16bf a = load_a_f32(X + (size_t)row0 * HH + kk, HH, lane);
    v16bf b = load_bt_bf16(WT + (size_t)col0 * HH + kk, HH, lane);
    acc = wmma_bf16(a, b, acc);
  }
  int n = lane & 15, mh = lane >> 4;
#pragma unroll
  for (int r = 0; r < 8; ++r) {
    int row = row0 + r + 8 * mh;
    int col = col0 + n;
    float val = (acc[r] + bvec[col]) * scale;
    if (!vmode) {
      dst[(size_t)row * HH + col] = (__bf16)val;        // [B*N, H]
    } else {
      int b_ = row >> 8, nn = row & 255;                // vT: [B, H, N]
      dst[((size_t)b_ * HH + col) * NN + nn] = (__bf16)val;
    }
  }
}

// ---------------------------------------------------------------------------
// K3: content scores  S[b,h,q,k] = q16 @ k^T + bias
// grid (B*12, N/16), block 256 (8 waves, 2 k-tiles each)
// ---------------------------------------------------------------------------
__global__ __launch_bounds__(256) void k_content_scores(
    const __bf16* __restrict__ qb, const __bf16* __restrict__ kb,
    const float* __restrict__ bias, float* __restrict__ S) {
  int bh = blockIdx.x;
  int b = bh / NHEADS, h = bh - b * NHEADS;
  int qt = blockIdx.y;
  int lane = threadIdx.x & 31, w = threadIdx.x >> 5;
  const __bf16* abase = qb + ((size_t)b * NN + qt * 16) * HH + h * DD;
  for (int t = w; t < 16; t += 8) {
    int kn0 = t * 16;
    const __bf16* bbase = kb + ((size_t)b * NN + kn0) * HH + h * DD;
    v8f acc = {};
#pragma unroll
    for (int kk = 0; kk < DD; kk += 32)
      acc = wmma_bf16(load_a_bf16(abase + kk, HH, lane),
                      load_bt_bf16(bbase + kk, HH, lane), acc);
    int n = lane & 15, mh = lane >> 4;
#pragma unroll
    for (int r = 0; r < 8; ++r) {
      int qq = qt * 16 + r + 8 * mh;
      size_t idx = (((size_t)bh * NN) + qq) * NN + kn0 + n;
      S[idx] = acc[r] + bias[idx];
    }
  }
}

// ---------------------------------------------------------------------------
// LayerNorm of one 64-row x 128-col chunk of paths into LDS (bf16).
// 4 threads per row, 32 cols each; shfl reduce across the 4-lane group.
// ---------------------------------------------------------------------------
__device__ __forceinline__ void ln_chunk(const float* __restrict__ src,
                                         __bf16* __restrict__ pn,
                                         const float* __restrict__ g,
                                         const float* __restrict__ bb, int tid) {
  int r = tid >> 2, sub = tid & 3;
  const float* x = src + (size_t)r * PP + sub * 32;
  float vals[32];
  float s = 0.f, s2 = 0.f;
#pragma unroll
  for (int i = 0; i < 32; ++i) {
    float v = x[i];
    vals[i] = v;
    s += v;
    s2 += v * v;
  }
  s  += __shfl_xor(s, 1, 32);  s  += __shfl_xor(s, 2, 32);
  s2 += __shfl_xor(s2, 1, 32); s2 += __shfl_xor(s2, 2, 32);
  float mean = s * (1.f / PP);
  float var  = s2 * (1.f / PP) - mean * mean;
  float rstd = rsqrtf(var + EPS);
#pragma unroll
  for (int i = 0; i < 32; ++i) {
    int col = sub * 32 + i;
    pn[r * PP + col] = (__bf16)((vals[i] - mean) * rstd * g[col] + bb[col]);
  }
}

// ---------------------------------------------------------------------------
// K4: fused per-(b,q) kernel.
//  pass1: LN(paths chunk) -> struct_k chunk (WMMA) -> struct scores (WMMA)+S
//  softmax over 256
//  pass2: LN again (L2 hit) -> struct_v chunk (WMMA, transposed in LDS)
//         -> out_s = probs @ struct_v (WMMA)
// grid 2048 blocks (b*256+q), block 256 (8 waves). 48KB static LDS.
// ---------------------------------------------------------------------------
__global__ __launch_bounds__(256) void k_attn_struct(
    const float* __restrict__ paths, const __bf16* __restrict__ qb,
    const float* __restrict__ S,
    const __bf16* __restrict__ WskT, const float* __restrict__ bsk,
    const __bf16* __restrict__ WsvT, const float* __restrict__ bsv,
    const float* __restrict__ g_path, const float* __restrict__ b_path,
    __bf16* __restrict__ P, float* __restrict__ outbuf) {
  __shared__ __bf16 pn[64 * PP];        // 16 KB  LN'd paths chunk
  __shared__ __bf16 skbuf[64 * DD];     //  8 KB  struct_k chunk / struct_v^T chunk
  __shared__ float  scores[16 * NN];    // 16 KB
  __shared__ __bf16 probs[16 * NN];     //  8 KB

  int b = blockIdx.x >> 8, qrow = blockIdx.x & 255;
  int tid = threadIdx.x, lane = tid & 31, w = tid >> 5;
  const float*  prow = paths + (((size_t)b * NN + qrow) * NN) * PP;
  const __bf16* qh   = qb + ((size_t)b * NN + qrow) * HH;
  int n = lane & 15, mh = lane >> 4;

  // ---------------- pass 1: scores ----------------
  for (int c = 0; c < 4; ++c) {
    if (c < 3)  // prefetch next 32KB chunk (global_prefetch_b8)
      __builtin_prefetch(prow + (size_t)(c + 1) * 64 * PP + tid * 32, 0, 1);
    ln_chunk(prow + (size_t)c * 64 * PP, pn, g_path, b_path, tid);
    __syncthreads();
    // struct_k chunk: pn(64x128) @ Wsk(128x64); 16 tiles, 2 per wave
#pragma unroll
    for (int ti = 0; ti < 2; ++ti) {
      int t = w * 2 + ti, tm = t >> 2, tn = t & 3;
      v8f acc = {};
#pragma unroll
      for (int kk = 0; kk < PP; kk += 32)
        acc = wmma_bf16(load_a_bf16(pn + tm * 16 * PP + kk, PP, lane),
                        load_bt_bf16(WskT + tn * 16 * PP + kk, PP, lane), acc);
#pragma unroll
      for (int r = 0; r < 8; ++r) {
        int kn = tm * 16 + r + 8 * mh, d = tn * 16 + n;
        skbuf[kn * DD + d] = (__bf16)(acc[r] + bsk[d]);
      }
    }
    __syncthreads();
    // struct scores: Qh(16x64) @ sk^T; waves 0-3 each one kn-tile
    if (w < 4) {
      v8f acc = {};
#pragma unroll
      for (int kk = 0; kk < DD; kk += 32)
        acc = wmma_bf16(load_a_qheads(qh, kk, lane),
                        load_bt_bf16(skbuf + w * 16 * DD + kk, DD, lane), acc);
#pragma unroll
      for (int r = 0; r < 8; ++r) {
        int h = r + 8 * mh, kn = c * 64 + w * 16 + n;
        float sc = 0.f;
        if (h < NHEADS)
          sc = acc[r] + S[(((size_t)(b * NHEADS + h)) * NN + qrow) * NN + kn];
        scores[h * NN + kn] = sc;
      }
    }
    __syncthreads();
  }

  // ---------------- softmax (rows = heads) ----------------
  {
    int row = tid >> 4, lc = tid & 15;
    float mx = -1e30f;
#pragma unroll
    for (int i = 0; i < 16; ++i) mx = fmaxf(mx, scores[row * NN + lc * 16 + i]);
#pragma unroll
    for (int m = 1; m < 16; m <<= 1) mx = fmaxf(mx, __shfl_xor(mx, m, 32));
    float e[16], sum = 0.f;
#pragma unroll
    for (int i = 0; i < 16; ++i) {
      e[i] = __expf(scores[row * NN + lc * 16 + i] - mx);
      sum += e[i];
    }
#pragma unroll
    for (int m = 1; m < 16; m <<= 1) sum += __shfl_xor(sum, m, 32);
    float inv = 1.f / sum;
#pragma unroll
    for (int i = 0; i < 16; ++i) {
      float p = e[i] * inv;
      int kn = lc * 16 + i;
      probs[row * NN + kn] = (__bf16)p;
      if (row < NHEADS)
        P[(((size_t)(b * NHEADS + row)) * NN + qrow) * NN + kn] = (__bf16)p;
    }
  }
  __syncthreads();

  // ---------------- pass 2: struct output ----------------
  v8f oacc = {};
  for (int c = 0; c < 4; ++c) {
    ln_chunk(prow + (size_t)c * 64 * PP, pn, g_path, b_path, tid);
    __syncthreads();
    // struct_v chunk: pn @ Wsv, stored TRANSPOSED: skbuf[d][kn_local]
#pragma unroll
    for (int ti = 0; ti < 2; ++ti) {
      int t = w * 2 + ti, tm = t >> 2, tn = t & 3;
      v8f acc = {};
#pragma unroll
      for (int kk = 0; kk < PP; kk += 32)
        acc = wmma_bf16(load_a_bf16(pn + tm * 16 * PP + kk, PP, lane),
                        load_bt_bf16(WsvT + tn * 16 * PP + kk, PP, lane), acc);
#pragma unroll
      for (int r = 0; r < 8; ++r) {
        int kn = tm * 16 + r + 8 * mh, d = tn * 16 + n;
        skbuf[d * 64 + kn] = (__bf16)(acc[r] + bsv[d]);
      }
    }
    __syncthreads();
    // out_s += probs[:, chunk](16x64) @ sv_chunk(64x64); waves 0-3 -> d-tiles
    if (w < 4) {
#pragma unroll
      for (int kk = 0; kk < 64; kk += 32)
        oacc = wmma_bf16(load_a_bf16(probs + c * 64 + kk, NN, lane),
                         load_bt_bf16(skbuf + w * 16 * 64 + kk, 64, lane), oacc);
    }
    __syncthreads();
  }
  if (w < 4) {
#pragma unroll
    for (int r = 0; r < 8; ++r) {
      int h = r + 8 * mh;
      if (h < NHEADS)
        outbuf[((size_t)b * NN + qrow) * HH + h * DD + w * 16 + n] = oacc[r];
    }
  }
}

// ---------------------------------------------------------------------------
// K5: content output  out[b,q,h*64+d] += P(16x256) @ v(256x64)
// grid (B*12, N/16), block 128 (4 waves -> 4 d-tiles)
// ---------------------------------------------------------------------------
__global__ __launch_bounds__(128) void k_content_out(
    const __bf16* __restrict__ P, const __bf16* __restrict__ vT,
    float* __restrict__ outbuf) {
  int bh = blockIdx.x;
  int b = bh / NHEADS, h = bh - b * NHEADS;
  int qt = blockIdx.y;
  int lane = threadIdx.x & 31, w = threadIdx.x >> 5;
  const __bf16* abase = P + ((size_t)bh * NN + qt * 16) * NN;
  const __bf16* bbase = vT + ((size_t)b * HH + h * DD + w * 16) * NN;
  v8f acc = {};
#pragma unroll
  for (int kk = 0; kk < NN; kk += 32)
    acc = wmma_bf16(load_a_bf16(abase + kk, NN, lane),
                    load_bt_bf16(bbase + kk, NN, lane), acc);
  int n = lane & 15, mh = lane >> 4;
#pragma unroll
  for (int r = 0; r < 8; ++r) {
    int q = qt * 16 + r + 8 * mh;
    size_t idx = ((size_t)b * NN + q) * HH + h * DD + w * 16 + n;
    outbuf[idx] += acc[r];
  }
}

// ---------------------------------------------------------------------------
// K6: Y = relu(out @ Wo + bo) + nodes     grid (64,12), block 256
// ---------------------------------------------------------------------------
__global__ __launch_bounds__(256) void k_out_proj(
    const float* __restrict__ X, const __bf16* __restrict__ WoT,
    const float* __restrict__ bo, const float* __restrict__ nodes,
    float* __restrict__ Y) {
  int lane = threadIdx.x & 31, w = threadIdx.x >> 5;
  int row0 = blockIdx.x * 32 + (w & 1) * 16;
  int col0 = blockIdx.y * 64 + (w >> 1) * 16;
  v8f acc = {};
#pragma unroll 4
  for (int kk = 0; kk < HH; kk += 32) {
    v16bf a = load_a_f32(X + (size_t)row0 * HH + kk, HH, lane);
    v16bf b = load_bt_bf16(WoT + (size_t)col0 * HH + kk, HH, lane);
    acc = wmma_bf16(a, b, acc);
  }
  int n = lane & 15, mh = lane >> 4;
#pragma unroll
  for (int r = 0; r < 8; ++r) {
    int row = row0 + r + 8 * mh, col = col0 + n;
    size_t idx = (size_t)row * HH + col;
    Y[idx] = fmaxf(acc[r] + bo[col], 0.f) + nodes[idx];
  }
}

// ---------------------------------------------------------------------------
// K7: final row LayerNorm over H=768 -> d_out
// ---------------------------------------------------------------------------
__global__ __launch_bounds__(256) void k_final_ln(
    const float* __restrict__ Y, const float* __restrict__ g,
    const float* __restrict__ bb, float* __restrict__ out) {
  __shared__ float rs[8], rs2[8];
  int row = blockIdx.x, tid = threadIdx.x, lane = tid & 31, w = tid >> 5;
  const float* x = Y + (size_t)row * HH;
  float v[3], s = 0.f, s2 = 0.f;
#pragma unroll
  for (int i = 0; i < 3; ++i) {
    v[i] = x[tid + 256 * i];
    s += v[i];
    s2 += v[i] * v[i];
  }
#pragma unroll
  for (int m = 1; m < 32; m <<= 1) {
    s += __shfl_xor(s, m, 32);
    s2 += __shfl_xor(s2, m, 32);
  }
  if (lane == 0) { rs[w] = s; rs2[w] = s2; }
  __syncthreads();
  float ts = 0.f, ts2 = 0.f;
#pragma unroll
  for (int i = 0; i < 8; ++i) { ts += rs[i]; ts2 += rs2[i]; }
  float mean = ts * (1.f / HH);
  float var  = ts2 * (1.f / HH) - mean * mean;
  float rstd = rsqrtf(var + EPS);
#pragma unroll
  for (int i = 0; i < 3; ++i) {
    int col = tid + 256 * i;
    out[(size_t)row * HH + col] = (v[i] - mean) * rstd * g[col] + bb[col];
  }
}

// ---------------------------------------------------------------------------
// Host launcher
// ---------------------------------------------------------------------------
extern "C" void kernel_launch(void* const* d_in, const int* in_sizes, int n_in,
                              void* d_out, int out_size, void* d_ws, size_t ws_size,
                              hipStream_t stream) {
  const float* nodes  = (const float*)d_in[0];
  const float* bias   = (const float*)d_in[1];
  const float* paths  = (const float*)d_in[2];
  const float* Wq     = (const float*)d_in[3];
  const float* bq     = (const float*)d_in[4];
  const float* Wk     = (const float*)d_in[5];
  const float* bk     = (const float*)d_in[6];
  const float* Wv     = (const float*)d_in[7];
  const float* bv     = (const float*)d_in[8];
  const float* Wsk    = (const float*)d_in[9];
  const float* bsk    = (const float*)d_in[10];
  const float* Wsv    = (const float*)d_in[11];
  const float* bsv    = (const float*)d_in[12];
  const float* Wo     = (const float*)d_in[13];
  const float* bo     = (const float*)d_in[14];
  const float* g_path = (const float*)d_in[15];
  const float* b_path = (const float*)d_in[16];
  const float* g_out  = (const float*)d_in[17];
  const float* b_out  = (const float*)d_in[18];

  // ---- workspace carve-up (256B aligned) ----
  char* ws = (char*)d_ws;
  size_t off = 0;
  auto carve = [&](size_t bytes) -> char* {
    char* p = ws + off;
    off += (bytes + 255) & ~(size_t)255;
    return p;
  };
  __bf16* WqT  = (__bf16*)carve((size_t)HH * HH * 2);
  __bf16* WkT  = (__bf16*)carve((size_t)HH * HH * 2);
  __bf16* WvT  = (__bf16*)carve((size_t)HH * HH * 2);
  __bf16* WoT  = (__bf16*)carve((size_t)HH * HH * 2);
  __bf16* WskT = (__bf16*)carve((size_t)PP * DD * 2);
  __bf16* WsvT = (__bf16*)carve((size_t)PP * DD * 2);
  __bf16* qb   = (__bf16*)carve((size_t)MROWS * HH * 2);
  __bf16* kb   = (__bf16*)carve((size_t)MROWS * HH * 2);
  __bf16* vT   = (__bf16*)carve((size_t)BB * HH * NN * 2);
  float*  S    = (float*) carve((size_t)BB * NHEADS * NN * NN * 4);
  __bf16* Pbuf = (__bf16*)carve((size_t)BB * NHEADS * NN * NN * 2);
  float*  outb = (float*) carve((size_t)MROWS * HH * 4);
  float*  Y    = (float*) carve((size_t)MROWS * HH * 4);
  (void)ws_size; (void)in_sizes; (void)n_in; (void)out_size;

  // K0: weight transposes -> bf16
  k_transpose_bf16<<<(HH * HH + 255) / 256, 256, 0, stream>>>(Wq, WqT, HH, HH);
  k_transpose_bf16<<<(HH * HH + 255) / 256, 256, 0, stream>>>(Wk, WkT, HH, HH);
  k_transpose_bf16<<<(HH * HH + 255) / 256, 256, 0, stream>>>(Wv, WvT, HH, HH);
  k_transpose_bf16<<<(HH * HH + 255) / 256, 256, 0, stream>>>(Wo, WoT, HH, HH);
  k_transpose_bf16<<<(PP * DD + 255) / 256, 256, 0, stream>>>(Wsk, WskT, PP, DD);
  k_transpose_bf16<<<(PP * DD + 255) / 256, 256, 0, stream>>>(Wsv, WsvT, PP, DD);

  // K1: projections (q pre-scaled by D^-0.5 = 0.125; v stored transposed)
  dim3 gproj(MROWS / 32, HH / 64);
  k_gemm_qkv<<<gproj, 256, 0, stream>>>(nodes, WqT, bq, qb, 0.125f, 0);
  k_gemm_qkv<<<gproj, 256, 0, stream>>>(nodes, WkT, bk, kb, 1.0f, 0);
  k_gemm_qkv<<<gproj, 256, 0, stream>>>(nodes, WvT, bv, vT, 1.0f, 1);

  // K3: content scores + bias
  k_content_scores<<<dim3(BB * NHEADS, NN / 16), 256, 0, stream>>>(qb, kb, bias, S);

  // K4: fused LN(paths) + struct-k/v + struct scores + softmax + struct out
  k_attn_struct<<<BB * NN, 256, 0, stream>>>(paths, qb, S, WskT, bsk, WsvT, bsv,
                                             g_path, b_path, Pbuf, outb);

  // K5: content attention output (adds into outb)
  k_content_out<<<dim3(BB * NHEADS, NN / 16), 128, 0, stream>>>(Pbuf, vT, outb);

  // K6: output projection + relu + residual
  k_out_proj<<<gproj, 256, 0, stream>>>(outb, WoT, bo, nodes, Y);

  // K7: final LayerNorm -> d_out
  k_final_ln<<<MROWS, 256, 0, stream>>>(Y, g_out, b_out, (float*)d_out);
}
